// PINN_Euler_57174604644935
// MI455X (gfx1250) — matrix operand order repaired and checked by reading, not compiled, not run on hardware
//
#include <hip/hip_runtime.h>

typedef _Float16 half8  __attribute__((ext_vector_type(8)));
typedef _Float16 half16 __attribute__((ext_vector_type(16)));
typedef float    float8 __attribute__((ext_vector_type(8)));
typedef unsigned int u32x4 __attribute__((ext_vector_type(4)));
typedef int      i32x4 __attribute__((ext_vector_type(4)));
typedef int      i32x8 __attribute__((ext_vector_type(8)));

#define EPSF 1e-10f
#define GM1  0.4f            // gamma - 1

// ---------------------------------------------------------------------------
// Workspace layout:
//   [0, 32)        : float acc[8]   (0..3 = residual sq-sums, 4 = ic sq-sum)
//   [128, ...)     : packed f16 weights:
//                      5 hidden layers * 65536 halves  (B-fragment order)
//                      + 4096 halves output layer (256x4 padded to 256x16)
// ---------------------------------------------------------------------------
#define HID_HALVES   (5 * 65536)
#define OUT_HALVES   4096
#define PACK_HALVES  (HID_HALVES + OUT_HALVES)

// Dynamic LDS layout (bytes):
//   [0, 65536)        activation buffer 0 (128 rows x 256 f16)
//   [65536, 131072)   activation buffer 1
//   [131072, 262144)  staged layer weights (65536 halves, fragment order)
//   [262144, 264192)  outres: 128 x 4 f32
#define SMEM_BYTES   (2 * 65536 + 131072 + 2048)

struct PackArgs {
    const float* W[6];       // W1..W5 (256x256), W6 (256x4)
    _Float16*    dst;
};

// Pre-swizzle weights into per-lane WMMA B-fragment layout.
// Fragment (kt, nt): lane L holds column N = nt*16 + (L&15),
// halves e=0..15 are K = kt*32 + e + (L>=16 ? 16 : 0).
__global__ __launch_bounds__(256) void pack_weights(PackArgs P) {
    int idx = blockIdx.x * 256 + threadIdx.x;
    if (idx < HID_HALVES) {
        int l    = idx >> 16;
        int off  = idx & 65535;
        int kt   = off >> 13;
        int nt   = (off >> 9) & 15;
        int lane = (off >> 4) & 31;
        int e    = off & 15;
        int n    = nt * 16 + (lane & 15);
        int k    = kt * 32 + e + ((lane & 16) ? 16 : 0);
        P.dst[idx] = (_Float16)P.W[l][k * 256 + n];
    } else if (idx < PACK_HALVES) {
        int off  = idx - HID_HALVES;
        int kt   = off >> 9;
        int lane = (off >> 4) & 31;
        int e    = off & 15;
        int n    = lane & 15;
        int k    = kt * 32 + e + ((lane & 16) ? 16 : 0);
        float v  = (n < 4) ? P.W[5][k * 4 + n] : 0.0f;
        P.dst[idx] = (_Float16)v;
    }
}

struct MlpArgs {
    const float*    X;        // points: stride 3 (res) or 7 (ic)
    const float*    W0;       // 3x256
    const float*    b0;       // 256
    const _Float16* packH;    // packed hidden weights
    const float*    bh[5];    // hidden biases
    const _Float16* packO;    // packed output weights (padded)
    const float*    b6;       // 4
    float*          acc;      // reduction accumulators
};

// --- Weight staging: split issue/wait so the TDM overlaps compute. ---------
// Issue: wave 0 posts one tensor_load_to_lds (2D tile, 8192 x 8 f16,
// contiguous 131072 B). Wait: wave 0 drains TENSORcnt; caller barriers.
__device__ __forceinline__ void stage_issue(const _Float16* src,
                                            _Float16* wlds, int tid) {
#if __has_builtin(__builtin_amdgcn_tensor_load_to_lds)
    if (tid < 32) {
        unsigned long long ga = (unsigned long long)(uintptr_t)src;
        unsigned int ldsoff   = (unsigned int)(uintptr_t)wlds;
        // D# group0: count=1 (valid), lds_addr, global_addr[56:0], type=2.
        u32x4 g0 = { 1u, ldsoff,
                     (unsigned int)(ga & 0xffffffffu),
                     (unsigned int)(((ga >> 32) & 0x1ffffffu) | 0x80000000u) };
        // D# group1: wg_mask=0, data_size=1 (2B), tensor 8192x8,
        // tile 8192x8, dim0 stride 8192 (fully contiguous copy).
        i32x8 g1 = { 0x00010000, 0x20000000, 0x00080000, 0x20000000,
                     8, 8192, 0, 0 };
        i32x4 gz = { 0, 0, 0, 0 };
#if defined(__clang_major__) && (__clang_major__ >= 23)
        i32x8 gz8 = { 0, 0, 0, 0, 0, 0, 0, 0 };
        __builtin_amdgcn_tensor_load_to_lds(g0, g1, gz, gz, gz8, 0);
#else
        __builtin_amdgcn_tensor_load_to_lds(g0, g1, gz, gz, 0);
#endif
    }
#endif
}

__device__ __forceinline__ void stage_wait(const _Float16* src,
                                           _Float16* wlds, int tid) {
#if __has_builtin(__builtin_amdgcn_tensor_load_to_lds)
    if (tid < 32) __builtin_amdgcn_s_wait_tensorcnt(0);
#else
    // Fallback: synchronous cooperative copy, 16B per thread per iteration.
    const u32x4* s = (const u32x4*)src;
    u32x4*       d = (u32x4*)wlds;
    for (int i = tid; i < 8192; i += 256) d[i] = s[i];
#endif
}

// MODE 0: residual pass. 32 points/block, 4 streams (primal, d/dt, d/dx, d/dy)
//         -> 128 GEMM rows.
// MODE 1: IC pass. 128 points/block, 1 stream -> 128 GEMM rows.
template <int MODE>
__global__ __launch_bounds__(256) void pinn_mlp(MlpArgs A) {
    extern __shared__ char smem[];
    _Float16* __restrict__ buf0 = (_Float16*)smem;          // 128x256 f16
    _Float16* __restrict__ buf1 = buf0 + 128 * 256;         // 128x256 f16
    _Float16* __restrict__ wlds = buf1 + 128 * 256;         // 65536 halves
    float*    outres = (float*)(smem + 2 * 65536 + 131072); // 128x4 f32

    const int tid    = threadIdx.x;
    const int lane   = tid & 31;
    const int w      = tid >> 5;
    const int laneHi = lane >> 4;     // 0: lanes 0-15, 1: lanes 16-31
    const int col16  = lane & 15;
    const int rowbase = w * 16;

    const int  npts   = (MODE == 0) ? 32 : 128;
    const int  stride = (MODE == 0) ? 3 : 7;
    const long base   = (long)blockIdx.x * npts;

    // Kick off layer-0 weight DMA; it overlaps the input-layer scalar phase.
    stage_issue(A.packH, wlds, tid);

    // ---------------- input layer (3 -> 256), scalar math -------------------
    {
        const int n = tid;
        const float w0 = A.W0[n], w1 = A.W0[256 + n], w2 = A.W0[512 + n];
        const float bb = A.b0[n];
        for (int p = 0; p < npts; ++p) {
            const float* xp = A.X + (base + p) * stride;
            float t = xp[0], x = xp[1], y = xp[2];
            float a = tanhf(w0 * t + w1 * x + w2 * y + bb);
            buf0[p * 256 + n] = (_Float16)a;
            if (MODE == 0) {
                float s = 1.0f - a * a;   // tanh' at primal
                buf0[(32 + p) * 256 + n] = (_Float16)(w0 * s);
                buf0[(64 + p) * 256 + n] = (_Float16)(w1 * s);
                buf0[(96 + p) * 256 + n] = (_Float16)(w2 * s);
            }
        }
    }
    __syncthreads();

    // ---------------- 5 hidden layers (256 -> 256) via WMMA -----------------
    // Fully unrolled so the ping-pong buffers are compile-time pointers
    // (keeps all activation traffic on DS ops, not FLAT).
#pragma unroll 5
    for (int l = 0; l < 5; ++l) {
        _Float16* __restrict__ inb  = (l & 1) ? buf1 : buf0;
        _Float16* __restrict__ outb = (l & 1) ? buf0 : buf1;

        // Weights for this layer must be resident in LDS.
        stage_wait(A.packH + (size_t)l * 65536, wlds, tid);
        __syncthreads();

        // Load this wave's A fragments once (reused across all 16 N-tiles).
        half16 afrag[8];
#pragma unroll
        for (int kt = 0; kt < 8; ++kt) {
            const int kbase = kt * 32 + laneHi * 16;
            const _Float16* rp = inb + (rowbase + col16) * 256 + kbase;
            half8 lo = *(const half8*)rp;
            half8 hi = *(const half8*)(rp + 8);
            half16 a;
#pragma unroll
            for (int e = 0; e < 8; ++e) { a[e] = lo[e]; a[e + 8] = hi[e]; }
            afrag[kt] = a;
        }

        const _Float16* pw = wlds + lane * 16;
        for (int nt = 0; nt < 16; ++nt) {
            float8 acc = {};
#pragma unroll
            for (int kt = 0; kt < 8; ++kt) {
                half16 bfrag = *(const half16*)(pw + (kt * 16 + nt) * 512);
                acc = __builtin_amdgcn_wmma_f32_16x16x32_f16(
                        false, afrag[kt], false, bfrag, (short)0, acc, false, false);
            }
            // C tile: lane holds col N=col16; VGPR r holds row r (+8 hi half).
            _Float16* cp = outb + (rowbase + laneHi * 8) * 256 + nt * 16 + col16;
#pragma unroll
            for (int r = 0; r < 8; ++r)
                cp[r * 256] = (_Float16)acc[r];
        }
        __syncthreads();

        // GEMM done reading wlds: start DMA of the next layer's weights now;
        // it overlaps the elementwise activation phase below.
        if (l < 4)
            stage_issue(A.packH + (size_t)(l + 1) * 65536, wlds, tid);

        // Elementwise: primal a = tanh(z + b); tangents dz *= (1 - a^2).
        {
            const float* bb = A.bh[l];
            const int n = tid;
            const float bn = bb[n];
            for (int p = 0; p < npts; ++p) {
                float a = tanhf((float)outb[p * 256 + n] + bn);
                outb[p * 256 + n] = (_Float16)a;
                if (MODE == 0) {
                    float s = 1.0f - a * a;
#pragma unroll
                    for (int st = 1; st < 4; ++st) {
                        int row = st * 32 + p;
                        float dz = (float)outb[row * 256 + n];
                        outb[row * 256 + n] = (_Float16)(dz * s);
                    }
                }
            }
        }
        __syncthreads();
    }

    // ---------------- output layer (256 -> 4, padded to 16) -----------------
    // After 5 ping-pongs the final activations live in buf1.
    {
        const _Float16* __restrict__ inb = buf1;
        const _Float16* pw = A.packO + lane * 16;
        float8 acc = {};
#pragma unroll
        for (int kt = 0; kt < 8; ++kt) {
            const int kbase = kt * 32 + laneHi * 16;
            const _Float16* rp = inb + (rowbase + col16) * 256 + kbase;
            half8 lo = *(const half8*)rp;
            half8 hi = *(const half8*)(rp + 8);
            half16 a;
#pragma unroll
            for (int e = 0; e < 8; ++e) { a[e] = lo[e]; a[e + 8] = hi[e]; }
            half16 bfrag = *(const half16*)(pw + (size_t)kt * 512);
            acc = __builtin_amdgcn_wmma_f32_16x16x32_f16(
                    false, a, false, bfrag, (short)0, acc, false, false);
        }
        if (col16 < 4) {
            const float bo = A.b6[col16];
            float* op = outres + (rowbase + laneHi * 8) * 4 + col16;
#pragma unroll
            for (int r = 0; r < 8; ++r)
                op[r * 4] = acc[r] + bo;
        }
    }
    __syncthreads();

    // ---------------- epilogue ----------------------------------------------
    if (MODE == 0) {
        if (tid < 32) {
            const int p = tid;
            const float r0 = outres[p * 4 + 0];
            const float m  = outres[p * 4 + 1];
            const float nq = outres[p * 4 + 2];
            const float E  = outres[p * 4 + 3];
            const float inv = 1.0f / (r0 + EPSF);
            const float u = m * inv, v = nq * inv;
            const float mu = m * u, nv = nq * v;
            const float pp = GM1 * (E - 0.5f * (mu + nv));

            float res[4];
            // u_t: U == ruve, so JVP along t is the tangent itself.
#pragma unroll
            for (int c = 0; c < 4; ++c) res[c] = outres[(32 + p) * 4 + c];
            // f_x = dF(r) . r_x
            {
                const float* d = &outres[(64 + p) * 4];
                float dr = d[0], dm = d[1], dn = d[2], dE = d[3];
                float du = (dm - u * dr) * inv;
                float dv = (dn - v * dr) * inv;
                float dmu = dm * u + m * du;
                float dnv = dn * v + nq * dv;
                float dp  = GM1 * (dE - 0.5f * (dmu + dnv));
                float druv = dr * u * v + r0 * (du * v + u * dv);
                res[0] += dm;
                res[1] += dmu + dp;
                res[2] += druv;
                res[3] += (dE + dp) * u + (E + pp) * du;
            }
            // g_y = dG(r) . r_y
            {
                const float* d = &outres[(96 + p) * 4];
                float dr = d[0], dm = d[1], dn = d[2], dE = d[3];
                float du = (dm - u * dr) * inv;
                float dv = (dn - v * dr) * inv;
                float dmu = dm * u + m * du;
                float dnv = dn * v + nq * dv;
                float dp  = GM1 * (dE - 0.5f * (dmu + dnv));
                float druv = dr * u * v + r0 * (du * v + u * dv);
                res[0] += dn;
                res[1] += druv;
                res[2] += dnv + dp;
                res[3] += (dE + dp) * v + (E + pp) * dv;
            }
#pragma unroll
            for (int c = 0; c < 4; ++c)
                atomicAdd(&A.acc[c], res[c] * res[c]);
        }
    } else {
        if (tid < 128) {
            const int p = tid;
            const float r0 = outres[p * 4 + 0];
            const float m  = outres[p * 4 + 1];
            const float nq = outres[p * 4 + 2];
            const float E  = outres[p * 4 + 3];
            const float inv = 1.0f / (r0 + EPSF);
            const float u = m * inv, v = nq * inv;
            const float pp = GM1 * (E - 0.5f * (m * u + nq * v));
            const float* xp = A.X + (base + p) * 7;
            float e0 = r0 - xp[3];
            float e1 = u  - xp[4];
            float e2 = v  - xp[5];
            float e3 = pp - xp[6];
            atomicAdd(&A.acc[4], e0 * e0 + e1 * e1 + e2 * e2 + e3 * e3);
        }
    }
}

__global__ void zero_acc(float* acc) {
    if (threadIdx.x < 8) acc[threadIdx.x] = 0.0f;
}

__global__ void finalize(const float* acc, float* out) {
    if (threadIdx.x == 0) {
        float le0 = acc[0] / 65536.0f;
        float le1 = acc[1] / 65536.0f;
        float le2 = acc[2] / 65536.0f;
        float le3 = acc[3] / 65536.0f;
        float lic = acc[4] / 32768.0f;
        float lE  = le0 + le1 + le2 + le3;
        out[0] = lE + 10.0f * lic;
        out[1] = le0; out[2] = le1; out[3] = le2; out[4] = le3;
        out[5] = lic;
    }
}

extern "C" void kernel_launch(void* const* d_in, const int* in_sizes, int n_in,
                              void* d_out, int out_size, void* d_ws, size_t ws_size,
                              hipStream_t stream) {
    const float* X_res = (const float*)d_in[0];
    const float* X_ic  = (const float*)d_in[1];
    const float* W[7];
    const float* B[7];
    for (int i = 0; i < 7; ++i) {
        W[i] = (const float*)d_in[2 + 2 * i];
        B[i] = (const float*)d_in[3 + 2 * i];
    }

    float*    acc  = (float*)d_ws;
    _Float16* pack = (_Float16*)((char*)d_ws + 128);

    (void)hipFuncSetAttribute((const void*)pinn_mlp<0>,
                              hipFuncAttributeMaxDynamicSharedMemorySize, SMEM_BYTES);
    (void)hipFuncSetAttribute((const void*)pinn_mlp<1>,
                              hipFuncAttributeMaxDynamicSharedMemorySize, SMEM_BYTES);

    zero_acc<<<1, 32, 0, stream>>>(acc);

    PackArgs P;
    for (int l = 0; l < 5; ++l) P.W[l] = W[1 + l];
    P.W[5] = W[6];
    P.dst  = pack;
    pack_weights<<<(PACK_HALVES + 255) / 256, 256, 0, stream>>>(P);

    MlpArgs Ar;
    Ar.X = X_res; Ar.W0 = W[0]; Ar.b0 = B[0];
    Ar.packH = pack;
    for (int l = 0; l < 5; ++l) Ar.bh[l] = B[1 + l];
    Ar.packO = pack + HID_HALVES; Ar.b6 = B[6];
    Ar.acc = acc;
    pinn_mlp<0><<<65536 / 32, 256, SMEM_BYTES, stream>>>(Ar);

    MlpArgs Ai = Ar;
    Ai.X = X_ic;
    pinn_mlp<1><<<32768 / 128, 256, SMEM_BYTES, stream>>>(Ai);

    finalize<<<1, 32, 0, stream>>>(acc, (float*)d_out);
}
